// MambaBlock_72636486910678
// MI455X (gfx1250) — matrix operand order repaired and checked by reading, not compiled, not run on hardware
//
#include <hip/hip_runtime.h>
#include <hip/hip_bf16.h>

typedef __attribute__((ext_vector_type(16))) _Float16 v16h;
typedef __attribute__((ext_vector_type(8)))  float    v8f;
typedef __attribute__((ext_vector_type(4)))  unsigned int tdm_g0_t;
typedef __attribute__((ext_vector_type(8)))  int          tdm_g1_t;
typedef __attribute__((ext_vector_type(4)))  int          tdm_g2_t;
typedef __attribute__((ext_vector_type(8)))  int          tdm_g4_t;

#define DM   1024
#define DI   2048
#define DS   16
#define DTR  64
#define NX   96      // dt_rank + 2*d_state
#define BSZ  4
#define SEQ  2048
#define MROWS (BSZ*SEQ)   // 8192

// ---------------------------------------------------------------------------
// fp32 -> fp16 convert
// ---------------------------------------------------------------------------
__global__ void cvt_f32_to_f16(const float* __restrict__ in,
                               _Float16* __restrict__ out, size_t n) {
    size_t i = (size_t)blockIdx.x * 256 + threadIdx.x;
    if (i < n) out[i] = (_Float16)in[i];
}

// ---------------------------------------------------------------------------
// WMMA f16 GEMM:  C[M,N] = A[M,K] * B[K,N] (+bias) (+softplus) (+f16 mirror)
// block = 256 threads = 8 waves; macro tile 128(M) x 64(N) x 32(K)
// wave grid 4(M) x 2(N); each wave computes 2x2 16x16 WMMA tiles.
// A tile is staged into LDS by the Tensor Data Mover (wave 0 issues a 2D
// tensor_load_to_lds with LDS padding 16B per 64B row -> stride 40 halves).
// Requires: M multiple of 128, K multiple of 32 (true for all call sites).
// ---------------------------------------------------------------------------
#define BM 128
#define BN 64
#define BK 32
#define LAS (BK + 8)   // LDS row stride (halves): 80B; TDM pad 4 DW per 16 DW
#define LBS (BK + 8)

__global__ __launch_bounds__(256)
void gemm_wmma_f16(const _Float16* __restrict__ A, int lda,
                   const _Float16* __restrict__ B, int ldb,
                   const float* __restrict__ bias,
                   float* __restrict__ C, int ldc,
                   _Float16* __restrict__ Ch, int ldh,
                   int M, int N, int K, int act)
{
    __shared__ _Float16 As[BM * LAS];
    __shared__ _Float16 Bs[BN * LBS];

    const int tid  = threadIdx.x;
    const int lane = tid & 31;
    const int wave = tid >> 5;
    const int wm   = wave & 3;      // 0..3 -> M offset wm*32
    const int wn   = wave >> 2;     // 0..1 -> N offset wn*32
    const int lm   = lane & 15;
    const int lh   = lane >> 4;

    const int bm = blockIdx.y * BM;
    const int bn = blockIdx.x * BN;

    v8f acc[2][2] = {};

    for (int kt = 0; kt < K; kt += BK) {
        // ---- stage A tile via Tensor Data Mover (wave 0 only) ----
        if (wave == 0) {
            unsigned lds_base = (unsigned)(size_t)(void*)As;
            unsigned long long ga =
                (unsigned long long)(size_t)(A + (size_t)bm * lda + kt);
            tdm_g0_t g0;
            g0[0] = 1u;                                   // count=1, user D#
            g0[1] = lds_base;                             // lds_addr
            g0[2] = (unsigned)ga;                         // global_addr[31:0]
            g0[3] = (unsigned)((ga >> 32) & 0x01FFFFFFu)  // global_addr[56:32]
                  | (2u << 30);                           // type=2 (image)
            tdm_g1_t g1;
            g1[0] = (int)((1u << 16)     // data_size = 2 bytes
                        | (1u << 20)     // pad_enable
                        | (3u << 22)     // pad_interval: every 16 DWORDs (64B)
                        | (3u << 25));   // pad_amount: 4 DWORDs (16B)
            g1[1] = (int)((unsigned)BK << 16);  // tensor_dim0 = 32 (lo16)
            g1[2] = (int)((unsigned)BM << 16);  // dim0 hi=0 | tensor_dim1 lo16
            g1[3] = (int)((unsigned)BK << 16);  // dim1 hi=0 | tile_dim0 = 32
            g1[4] = (int)BM;                    // tile_dim1=128 | tile_dim2=0
            g1[5] = lda;                        // tensor_dim0_stride (elements)
            g1[6] = 0;
            g1[7] = 0;
            tdm_g2_t gz4 = {0, 0, 0, 0};
            tdm_g4_t gz8 = {0, 0, 0, 0, 0, 0, 0, 0};
            __builtin_amdgcn_tensor_load_to_lds(g0, g1, gz4, gz4, gz8, 0);
        }

        // ---- stage B tile transposed to Bs[n][k] (overlaps with TDM) ----
        // thread covers k-pair kp=2*(tid&15), 4 n at ng=4*(tid>>4)
        {
            int kp = (tid & 15) * 2;
            int ng = (tid >> 4) * 4;
            const _Float16* src = B + (size_t)(kt + kp) * ldb + bn + ng;
            alignas(8) _Float16 rk[4], rk1[4];
            if (bn + ng + 3 < N) {
                *(float2*)rk  = *(const float2*)(src);
                *(float2*)rk1 = *(const float2*)(src + ldb);
                if (kt + BK < K)
                    __builtin_prefetch(src + (size_t)BK * ldb, 0, 1);
            } else {
                #pragma unroll
                for (int j = 0; j < 4; ++j) {
                    bool in = (bn + ng + j) < N;
                    rk[j]  = in ? src[j]       : (_Float16)0.0f;
                    rk1[j] = in ? src[ldb + j] : (_Float16)0.0f;
                }
            }
            #pragma unroll
            for (int j = 0; j < 4; ++j) {
                union { _Float16 h[2]; unsigned u; } p;
                p.h[0] = rk[j];
                p.h[1] = rk1[j];
                *(unsigned*)&Bs[(ng + j) * LBS + kp] = p.u;
            }
        }

        if (wave == 0) __builtin_amdgcn_s_wait_tensorcnt(0);
        __syncthreads();

        // ---- fragments per ISA VGPR layouts ----
        v16h afr[2], bfr[2];
        #pragma unroll
        for (int tm = 0; tm < 2; ++tm) {
            // A 16x32 f16: lane m=lm; K = 8*lh..8*lh+7 then 16+8*lh..+7
            const _Float16* ap = &As[(wm * 32 + tm * 16 + lm) * LAS + 8 * lh];
            union { v16h v; float4 q[2]; } u;
            u.q[0] = *(const float4*)(ap);
            u.q[1] = *(const float4*)(ap + 16);
            afr[tm] = u.v;
        }
        #pragma unroll
        for (int tn = 0; tn < 2; ++tn) {
            // B 32x16 f16: lane n=lm; K = 8*lh..8*lh+7 then 16+8*lh..+7
            const _Float16* bp = &Bs[(wn * 32 + tn * 16 + lm) * LBS + 8 * lh];
            union { v16h v; float4 q[2]; } u;
            u.q[0] = *(const float4*)(bp);
            u.q[1] = *(const float4*)(bp + 16);
            bfr[tn] = u.v;
        }
        #pragma unroll
        for (int tm = 0; tm < 2; ++tm)
            #pragma unroll
            for (int tn = 0; tn < 2; ++tn)
                acc[tm][tn] = __builtin_amdgcn_wmma_f32_16x16x32_f16(
                    false, afr[tm], false, bfr[tn],
                    (short)0, acc[tm][tn], false, false);
        __syncthreads();
    }

    // ---- epilogue: C layout lane n=lm, rows m = 8*lh + r ----
    #pragma unroll
    for (int tm = 0; tm < 2; ++tm)
        #pragma unroll
        for (int tn = 0; tn < 2; ++tn) {
            int n_g = bn + wn * 32 + tn * 16 + lm;
            if (n_g >= N) continue;
            float bv = bias ? bias[n_g] : 0.0f;
            #pragma unroll
            for (int r = 0; r < 8; ++r) {
                int m_g = bm + wm * 32 + tm * 16 + lh * 8 + r;
                if (m_g >= M) continue;
                float v = acc[tm][tn][r] + bv;
                if (act == 1) v = (v > 20.f) ? v : log1pf(__expf(v));  // softplus
                C[(size_t)m_g * ldc + n_g] = v;
                if (Ch) Ch[(size_t)m_g * ldh + n_g] = (_Float16)v;
            }
        }
}

// ---------------------------------------------------------------------------
// causal depthwise conv1d (width 4) + SiLU. Input = in_proj output
// (row stride 2*DI, x-half at cols [0,DI)). Writes f32 and f16 copies.
// ---------------------------------------------------------------------------
__global__ void conv_silu_kernel(const float* __restrict__ xin,
                                 const float* __restrict__ cw,
                                 const float* __restrict__ cb,
                                 float* __restrict__ xo,
                                 _Float16* __restrict__ xo16)
{
    size_t idx = (size_t)blockIdx.x * 256 + threadIdx.x;  // over MROWS*DI
    if (idx >= (size_t)MROWS * DI) return;
    int d = (int)(idx % DI);
    size_t bt = idx / DI;           // b*SEQ + t
    int t = (int)(bt % SEQ);
    const float* w = cw + d * 4;
    float acc = cb[d];
    #pragma unroll
    for (int k = 0; k < 4; ++k) {
        int tt = t - 3 + k;
        if (tt >= 0) acc += w[k] * xin[(bt - (size_t)(3 - k)) * (2 * DI) + d];
    }
    float s = acc / (1.f + __expf(-acc));   // silu
    xo[bt * DI + d]   = s;
    xo16[bt * DI + d] = (_Float16)s;
}

// ---------------------------------------------------------------------------
// selective scan: one thread per (b, d); h[16] in registers; B_t/C_t staged
// in LDS. Fuses D-skip, SiLU(res) gate, and f16 conversion of output.
// grid = (DI/256, BSZ), block = 256
// ---------------------------------------------------------------------------
__global__ __launch_bounds__(256)
void scan_kernel(const float* __restrict__ dtf,    // [MROWS, DI] softplus(dt)
                 const float* __restrict__ xconv,  // [MROWS, DI]
                 const float* __restrict__ xdbl,   // [MROWS, NX] (B at 64, C at 80)
                 const float* __restrict__ inproj, // [MROWS, 2*DI] (res at +DI)
                 const float* __restrict__ A_log,  // [DI, DS]
                 const float* __restrict__ Dp,     // [DI]
                 _Float16* __restrict__ yg)        // [MROWS, DI]
{
    const int b = blockIdx.y;
    const int d = blockIdx.x * 256 + threadIdx.x;

    float Aa[DS];
    #pragma unroll
    for (int s = 0; s < DS; ++s) Aa[s] = -__expf(A_log[d * DS + s]);
    const float Dv = Dp[d];
    float h[DS] = {};

    __shared__ float sB[DS], sC[DS];

    for (int t = 0; t < SEQ; ++t) {
        size_t bt = (size_t)b * SEQ + t;
        __syncthreads();                        // protect previous-iter reads
        if (threadIdx.x < 32) {
            const float* row = xdbl + bt * NX + DTR;
            if (threadIdx.x < DS) sB[threadIdx.x] = row[threadIdx.x];
            else                  sC[threadIdx.x - DS] = row[threadIdx.x];
        }
        __syncthreads();

        float dt = dtf[bt * DI + d];
        float xv = xconv[bt * DI + d];
        float y = 0.f;
        #pragma unroll
        for (int s = 0; s < DS; ++s) {
            float dA = __expf(dt * Aa[s]);
            h[s] = h[s] * dA + (dt * sB[s]) * xv;
            y += h[s] * sC[s];
        }
        y += Dv * xv;
        float r = inproj[bt * (2 * DI) + DI + d];
        float g = r / (1.f + __expf(-r));       // silu(res)
        yg[bt * DI + d] = (_Float16)(y * g);
    }
}

// ---------------------------------------------------------------------------
// launch
// ---------------------------------------------------------------------------
extern "C" void kernel_launch(void* const* d_in, const int* in_sizes, int n_in,
                              void* d_out, int out_size, void* d_ws, size_t ws_size,
                              hipStream_t stream) {
    (void)in_sizes; (void)n_in; (void)out_size; (void)ws_size;

    const float* x         = (const float*)d_in[0];
    const float* in_proj_w = (const float*)d_in[1];
    const float* in_proj_b = (const float*)d_in[2];
    const float* conv_w    = (const float*)d_in[3];
    const float* conv_b    = (const float*)d_in[4];
    const float* x_proj_w  = (const float*)d_in[5];
    const float* dt_proj_w = (const float*)d_in[6];
    const float* dt_proj_b = (const float*)d_in[7];
    const float* A_log     = (const float*)d_in[8];
    const float* Dp        = (const float*)d_in[9];
    const float* out_proj_w= (const float*)d_in[10];
    const float* out_proj_b= (const float*)d_in[11];
    float* out = (float*)d_out;

    // workspace carve-out (256B aligned)
    char* ws = (char*)d_ws;
    size_t off = 0;
    auto take = [&](size_t bytes) {
        char* p = ws + off;
        off += (bytes + 255) & ~(size_t)255;
        return p;
    };
    _Float16* x16     = (_Float16*)take((size_t)MROWS * DM * 2);
    _Float16* w_in16  = (_Float16*)take((size_t)DM * (2 * DI) * 2);
    _Float16* w_xp16  = (_Float16*)take((size_t)DI * NX * 2);
    _Float16* w_dt16  = (_Float16*)take((size_t)DTR * DI * 2);
    _Float16* w_out16 = (_Float16*)take((size_t)DI * DM * 2);
    float*    inproj  = (float*)   take((size_t)MROWS * (2 * DI) * 4);
    float*    xconv   = (float*)   take((size_t)MROWS * DI * 4);
    _Float16* xconv16 = (_Float16*)take((size_t)MROWS * DI * 2);
    float*    xdbl    = (float*)   take((size_t)MROWS * NX * 4);
    _Float16* dtraw16 = (_Float16*)take((size_t)MROWS * NX * 2);
    float*    dtfull  = (float*)   take((size_t)MROWS * DI * 4);
    _Float16* yg16    = (_Float16*)take((size_t)MROWS * DI * 2);

    auto cvt = [&](const float* src, _Float16* dst, size_t n) {
        cvt_f32_to_f16<<<dim3((unsigned)((n + 255) / 256)), dim3(256), 0, stream>>>(src, dst, n);
    };
    cvt(x,          x16,     (size_t)MROWS * DM);
    cvt(in_proj_w,  w_in16,  (size_t)DM * (2 * DI));
    cvt(x_proj_w,   w_xp16,  (size_t)DI * NX);
    cvt(dt_proj_w,  w_dt16,  (size_t)DTR * DI);
    cvt(out_proj_w, w_out16, (size_t)DI * DM);

    // in_proj: [8192,1024] x [1024,4096] + b
    gemm_wmma_f16<<<dim3((2 * DI) / BN, MROWS / BM), dim3(256), 0, stream>>>(
        x16, DM, w_in16, 2 * DI, in_proj_b, inproj, 2 * DI,
        (_Float16*)nullptr, 0, MROWS, 2 * DI, DM, 0);

    // depthwise conv + silu
    conv_silu_kernel<<<dim3((unsigned)(((size_t)MROWS * DI + 255) / 256)), dim3(256), 0, stream>>>(
        inproj, conv_w, conv_b, xconv, xconv16);

    // x_proj: [8192,2048] x [2048,96] ; f16 mirror feeds dt_proj (cols 0..63)
    gemm_wmma_f16<<<dim3((NX + BN - 1) / BN, MROWS / BM), dim3(256), 0, stream>>>(
        xconv16, DI, w_xp16, NX, (const float*)nullptr, xdbl, NX,
        dtraw16, NX, MROWS, NX, DI, 0);

    // dt_proj: [8192,64] x [64,2048] + b, softplus fused
    gemm_wmma_f16<<<dim3(DI / BN, MROWS / BM), dim3(256), 0, stream>>>(
        dtraw16, NX, w_dt16, DI, dt_proj_b, dtfull, DI,
        (_Float16*)nullptr, 0, MROWS, DI, DTR, 1);

    // selective scan + gate
    scan_kernel<<<dim3(DI / 256, BSZ), dim3(256), 0, stream>>>(
        dtfull, xconv, xdbl, inproj, A_log, Dp, yg16);

    // out_proj: [8192,2048] x [2048,1024] + b -> d_out (fp32)
    gemm_wmma_f16<<<dim3(DM / BN, MROWS / BM), dim3(256), 0, stream>>>(
        yg16, DI, w_out16, DM, out_proj_b, out, DM,
        (_Float16*)nullptr, 0, MROWS, DM, DI, 0);
}
